// LigerFusedLinearORPOLoss_4621384810879
// MI455X (gfx1250) — matrix operand order repaired and checked by reading, not compiled, not run on hardware
//
#include <hip/hip_runtime.h>
#include <hip/hip_bf16.h>

// ---------------- problem constants ----------------
#define H_DIM 2048
#define V_DIM 50257
#define NTOK  4096            // 4 * 1024 tokens
#define NSEQ  4
#define NSLAB 16
#define NT_TILES 3142         // ceil(50257 / 16)
#define TILES_PER_SLAB 197    // ceil(3142 / 16)
#define ORPO_BETA 0.1f

typedef __attribute__((ext_vector_type(16))) __bf16 v16bf;
typedef __attribute__((ext_vector_type(8)))  float  v8f;

#define WS_X_BYTES (16777216ull)                                    // 4096*2048 bf16 A-frags
#define WS_W_BYTES ((unsigned long long)NT_TILES * 64ull * 1024ull) // B-frags

__device__ __forceinline__ unsigned short f2bf(float f) {
  unsigned u = __float_as_uint(f);
  u += 0x7FFFu + ((u >> 16) & 1u);      // round-to-nearest-even
  return (unsigned short)(u >> 16);
}

// =====================================================================
// Kernel 1: pack x (fp32) into bf16 A-fragments in WMMA lane order.
// A-matrix 16x32 bf16 layout used here (contiguous k, not the ISA table order):
//   lanes 0-15 : M = lane,    K = {0..15}
//   lanes16-31 : M = lane-16, K = {16..31}
// Fragment f = (m_tile*64 + k_block); each fragment = 32 lanes * 16 bf16.
// =====================================================================
__global__ __launch_bounds__(256) void pack_x_kernel(const float* __restrict__ x,
                                                     unsigned short* __restrict__ xb,
                                                     float* __restrict__ tlog) {
  const int tid = threadIdx.x;
  if (blockIdx.x < 16) tlog[blockIdx.x * 256 + tid] = 0.0f;   // init target logits

  const int wave   = tid >> 5;
  const int lane   = tid & 31;
  const int frag   = blockIdx.x * 8 + wave;        // 0 .. 16383
  const int m_tile = frag >> 6;                    // 0 .. 255
  const int kb     = frag & 63;
  const int token  = m_tile * 16 + (lane & 15);
  const int half   = lane >> 4;

  unsigned short* dst = xb + ((size_t)frag * 32 + lane) * 16;
  const float* src = x + (size_t)token * H_DIM;
#pragma unroll
  for (int v = 0; v < 8; ++v) {
    const int k = kb * 32 + 2 * v + 16 * half;
    dst[2 * v + 0] = f2bf(src[k + 0]);
    dst[2 * v + 1] = f2bf(src[k + 1]);
  }
}

// =====================================================================
// Kernel 2: pack weight (fp32, row-major VxH) into bf16 B-fragments.
// B-matrix 32x16 (KxN) bf16 layout: lane l holds column N = l&15,
// K = 16*(l>>4) + j for j=0..15 (consecutive) -> one 64 B contiguous read.
// =====================================================================
__global__ __launch_bounds__(256) void pack_w_kernel(const float* __restrict__ w,
                                                     unsigned short* __restrict__ wf) {
  const int tid  = threadIdx.x;
  const int wave = tid >> 5;
  const int lane = tid & 31;
  const int frag = blockIdx.x * 8 + wave;          // 0 .. 201087
  const int nt   = frag >> 6;
  const int kb   = frag & 63;
  const int n    = lane & 15;
  const int half = lane >> 4;
  const int vrow = nt * 16 + n;
  const int ks   = kb * 32 + half * 16;

  unsigned short* dst = wf + ((size_t)frag * 32 + lane) * 16;
  if (vrow < V_DIM) {
    const float* src = w + (size_t)vrow * H_DIM + ks;
#pragma unroll
    for (int j = 0; j < 16; ++j) dst[j] = f2bf(src[j]);
  } else {
#pragma unroll
    for (int j = 0; j < 16; ++j) dst[j] = 0;
  }
}

// =====================================================================
// Templated MW=2 x NW-tile GEMM + online-softmax group (constant NWT
// keeps every accumulator index compile-time constant -> no scratch).
// =====================================================================
template <int NWT>
__device__ __forceinline__ void gemm_softmax_group(const char* a0base, const char* a1base,
                                                   const char* bbase, int nt, int mt0,
                                                   int lane,
                                                   const long long* __restrict__ target,
                                                   float rmax[2][8], float rsum[2][8],
                                                   float* __restrict__ tlog) {
#if defined(__HIP_DEVICE_COMPILE__)
  const int half  = lane >> 4;
  const int col16 = lane & 15;

  v8f acc[2][NWT];
#pragma unroll
  for (int m = 0; m < 2; ++m)
#pragma unroll
    for (int n = 0; n < NWT; ++n) { v8f z = {}; acc[m][n] = z; }

  for (int kb = 0; kb < 64; ++kb) {
    union { uint4 q[2]; v16bf v; } A0, A1;
    const uint4* p;
    p = (const uint4*)(a0base + (size_t)kb * 1024); A0.q[0] = p[0]; A0.q[1] = p[1];
    p = (const uint4*)(a1base + (size_t)kb * 1024); A1.q[0] = p[0]; A1.q[1] = p[1];
#pragma unroll
    for (int n = 0; n < NWT; ++n) {
      union { uint4 q[2]; v16bf v; } B;
      p = (const uint4*)(bbase + (size_t)n * 65536 + (size_t)kb * 1024);
      B.q[0] = p[0]; B.q[1] = p[1];
      acc[0][n] = __builtin_amdgcn_wmma_f32_16x16x32_bf16(false, A0.v, false, B.v, (short)0, acc[0][n], false, false);
      acc[1][n] = __builtin_amdgcn_wmma_f32_16x16x32_bf16(false, A1.v, false, B.v, (short)0, acc[1][n], false, false);
    }
  }

  // online-softmax update, tile order n = 0..NWT-1, m = 0..1
#pragma unroll
  for (int n = 0; n < NWT; ++n) {
    const int  col   = (nt + n) * 16 + col16;
    const bool colok = (col < V_DIM);
    const long long tcol = (long long)col;
#pragma unroll
    for (int m = 0; m < 2; ++m) {
#pragma unroll
      for (int j = 0; j < 8; ++j) {
        float lv = colok ? acc[m][n][j] : -__builtin_inff();
        float tm = lv;
#pragma unroll
        for (int off = 1; off < 16; off <<= 1) tm = fmaxf(tm, __shfl_xor(tm, off, 16));
        const float nm = fmaxf(rmax[m][j], tm);
        float ts = colok ? __expf(lv - nm) : 0.f;
#pragma unroll
        for (int off = 1; off < 16; off <<= 1) ts += __shfl_xor(ts, off, 16);
        rsum[m][j] = rsum[m][j] * __expf(rmax[m][j] - nm) + ts;
        rmax[m][j] = nm;
        const int token = (mt0 + m) * 16 + half * 8 + j;
        if (colok && target[token] == tcol) tlog[token] = acc[m][n][j];
      }
    }
  }
#endif
}

// =====================================================================
// Kernel 3 (FAST): pure-fragment GEMM + online softmax.
// grid = (16 token-blocks of 256) x (NSLAB slabs); 256 thr = 8 waves.
// Each wave: MW=2 M-tiles x NW=4 N-tiles -> 12x global_load_b128 per
// 8x v_wmma; A-side L2 traffic halved vs NW=2.
// =====================================================================
__global__ __launch_bounds__(256) void orpo_main_fast(const unsigned short* __restrict__ xb,
                                                      const unsigned short* __restrict__ wf,
                                                      const long long* __restrict__ target,
                                                      float* __restrict__ pmax,
                                                      float* __restrict__ psum,
                                                      float* __restrict__ tlog) {
#if defined(__HIP_DEVICE_COMPILE__)
  const int tid   = threadIdx.x;
  const int lane  = tid & 31;
  const int wave  = tid >> 5;
  const int half  = lane >> 4;
  const int col16 = lane & 15;

  const int m_block = blockIdx.x;                      // 0..15 (256 tokens)
  const int slab    = blockIdx.y;                      // 0..NSLAB-1
  const int mt0     = m_block * 16 + wave * 2;         // wave's two 16-token tiles
  const int nt0     = slab * TILES_PER_SLAB;
  const int nt1     = (nt0 + TILES_PER_SLAB < NT_TILES) ? nt0 + TILES_PER_SLAB : NT_TILES;

  const char* a0base = (const char*)xb + (size_t)(mt0 + 0) * 65536 + (size_t)lane * 32;
  const char* a1base = (const char*)xb + (size_t)(mt0 + 1) * 65536 + (size_t)lane * 32;

  float rmax[2][8], rsum[2][8];
#pragma unroll
  for (int m = 0; m < 2; ++m)
#pragma unroll
    for (int j = 0; j < 8; ++j) { rmax[m][j] = -__builtin_inff(); rsum[m][j] = 0.f; }

  int nt = nt0;
  for (; nt + 4 <= nt1; nt += 4) {
    __syncthreads();   // keep 8 waves in step -> WGP$ dedups shared B frags

    // prefetch next group's B fragments (up to 4 x 64 KiB, 256 B/thread each)
#pragma unroll
    for (int i = 0; i < 4; ++i)
      if (nt + 4 + i < nt1)
        __builtin_prefetch((const char*)wf + (size_t)(nt + 4 + i) * 65536 + (size_t)tid * 256, 0, 0);

    const char* bbase = (const char*)wf + (size_t)nt * 65536 + (size_t)lane * 32;
    gemm_softmax_group<4>(a0base, a1base, bbase, nt, mt0, lane, target, rmax, rsum, tlog);
  }

  const int rem = nt1 - nt;   // uniform per block
  if (rem > 0) {
    __syncthreads();
    const char* bbase = (const char*)wf + (size_t)nt * 65536 + (size_t)lane * 32;
    if (rem == 3)      gemm_softmax_group<3>(a0base, a1base, bbase, nt, mt0, lane, target, rmax, rsum, tlog);
    else if (rem == 2) gemm_softmax_group<2>(a0base, a1base, bbase, nt, mt0, lane, target, rmax, rsum, tlog);
    else               gemm_softmax_group<1>(a0base, a1base, bbase, nt, mt0, lane, target, rmax, rsum, tlog);
  }

  if (col16 == 0) {
#pragma unroll
    for (int m = 0; m < 2; ++m)
#pragma unroll
      for (int j = 0; j < 8; ++j) {
        const int token = (mt0 + m) * 16 + half * 8 + j;
        pmax[(size_t)token * NSLAB + slab] = rmax[m][j];
        psum[(size_t)token * NSLAB + slab] = rsum[m][j];
      }
  }
#endif
}

// =====================================================================
// Kernel 3 (FALLBACK, ws too small): fp32 weight staged through LDS
// with on-the-fly bf16 convert (double-buffered).  grid = (32, NSLAB).
// =====================================================================
#define B_STRIDE 72
#define B_BUF    (16 * B_STRIDE)

__device__ __forceinline__ void fill_b(unsigned short* smem, int buf,
                                       const float* __restrict__ W,
                                       int v_base, int kc) {
  const int t  = threadIdx.x;
  const int n  = t >> 4;
  const int k4 = (t & 15) << 2;
  const int vrow = v_base + n;
  float4 w;
  if (vrow < V_DIM) w = *(const float4*)&W[(size_t)vrow * H_DIM + (size_t)kc * 64 + k4];
  else              w = make_float4(0.f, 0.f, 0.f, 0.f);
  unsigned short* d = smem + buf * B_BUF + n * B_STRIDE + k4;
  d[0] = f2bf(w.x); d[1] = f2bf(w.y); d[2] = f2bf(w.z); d[3] = f2bf(w.w);
}

__global__ __launch_bounds__(256) void orpo_main_fallback(const float* __restrict__ W,
                                                          const long long* __restrict__ target,
                                                          const unsigned short* __restrict__ xb,
                                                          float* __restrict__ pmax,
                                                          float* __restrict__ psum,
                                                          float* __restrict__ tlog) {
#if defined(__HIP_DEVICE_COMPILE__)
  __shared__ unsigned short smem[2 * B_BUF];

  const int tid   = threadIdx.x;
  const int lane  = tid & 31;
  const int wave  = tid >> 5;
  const int half  = lane >> 4;
  const int col16 = lane & 15;

  const int m_block = blockIdx.x;
  const int slab    = blockIdx.y;
  const int m_tile  = m_block * 8 + wave;
  const int nt0 = slab * TILES_PER_SLAB;
  const int nt1 = (nt0 + TILES_PER_SLAB < NT_TILES) ? nt0 + TILES_PER_SLAB : NT_TILES;

  float run_max[8], run_sum[8];
#pragma unroll
  for (int j = 0; j < 8; ++j) { run_max[j] = -__builtin_inff(); run_sum[j] = 0.f; }

  for (int nt = nt0; nt < nt1; ++nt) {
    const int v_base = nt * 16;
    {
      const int vn = v_base + 16 + (tid >> 4);
      if (vn < V_DIM)
        __builtin_prefetch(&W[(size_t)vn * H_DIM + (size_t)(tid & 15) * 128], 0, 0);
    }

    v8f acc = {};
    fill_b(smem, 0, W, v_base, 0);
    for (int kc = 0; kc < 32; ++kc) {
      __syncthreads();
      if (kc + 1 < 32) fill_b(smem, (kc + 1) & 1, W, v_base, kc + 1);
      const unsigned short* buf = &smem[(kc & 1) * B_BUF];
#pragma unroll
      for (int s = 0; s < 2; ++s) {
        const int kb = kc * 2 + s;
        union { uint4 q[2]; v16bf v; } A, B;
        const uint4* ap = (const uint4*)(xb + ((size_t)(m_tile * 64 + kb) * 32 + lane) * 16);
        A.q[0] = ap[0]; A.q[1] = ap[1];
        const uint4* bp = (const uint4*)(buf + col16 * B_STRIDE + s * 32 + half * 16);
        B.q[0] = bp[0]; B.q[1] = bp[1];
        acc = __builtin_amdgcn_wmma_f32_16x16x32_bf16(false, A.v, false, B.v, (short)0, acc, false, false);
      }
    }

    const int  col   = v_base + col16;
    const bool colok = (col < V_DIM);
    const long long tcol = (long long)col;
#pragma unroll
    for (int j = 0; j < 8; ++j) {
      float lv = colok ? acc[j] : -__builtin_inff();
      float tm = lv;
#pragma unroll
      for (int off = 1; off < 16; off <<= 1) tm = fmaxf(tm, __shfl_xor(tm, off, 16));
      const float nm = fmaxf(run_max[j], tm);
      float ts = colok ? __expf(lv - nm) : 0.f;
#pragma unroll
      for (int off = 1; off < 16; off <<= 1) ts += __shfl_xor(ts, off, 16);
      run_sum[j] = run_sum[j] * __expf(run_max[j] - nm) + ts;
      run_max[j] = nm;
      const int token = m_block * 128 + wave * 16 + half * 8 + j;
      if (colok && target[token] == tcol) tlog[token] = acc[j];
    }
  }

  if (col16 == 0) {
#pragma unroll
    for (int j = 0; j < 8; ++j) {
      const int token = m_block * 128 + wave * 16 + half * 8 + j;
      pmax[(size_t)token * NSLAB + slab] = run_max[j];
      psum[(size_t)token * NSLAB + slab] = run_sum[j];
    }
  }
#endif
}

// =====================================================================
// Kernel 4: merge slab partials, per-sequence average, ORPO outputs.
// Deterministic fixed-order tree reduction (no float atomics).
// =====================================================================
__device__ __forceinline__ float log_sigmoid(float x) {
  return (x >= 0.f) ? -log1pf(__expf(-x)) : x - log1pf(__expf(x));
}

__global__ __launch_bounds__(256) void orpo_finalize_kernel(const long long* __restrict__ target,
                                                            const float* __restrict__ pmax,
                                                            const float* __restrict__ psum,
                                                            const float* __restrict__ tlog,
                                                            float* __restrict__ out) {
  __shared__ float red[256];
  __shared__ float seq_lp[NSEQ], seq_ct[NSEQ];
  const int tid = threadIdx.x;

  float acc_lp[NSEQ] = {0.f, 0.f, 0.f, 0.f};
  float acc_ct[NSEQ] = {0.f, 0.f, 0.f, 0.f};

  for (int t = tid; t < NTOK; t += 256) {
    float M = -__builtin_inff();
#pragma unroll 4
    for (int s = 0; s < NSLAB; ++s) M = fmaxf(M, pmax[(size_t)t * NSLAB + s]);
    float S = 0.f;
#pragma unroll 4
    for (int s = 0; s < NSLAB; ++s)
      S += psum[(size_t)t * NSLAB + s] * __expf(pmax[(size_t)t * NSLAB + s] - M);
    const float lp = tlog[t] - (M + __logf(S));
    const float mk = (target[t] != -100LL) ? 1.f : 0.f;
    const int b = t >> 10;
    acc_lp[b] += lp * mk;
    acc_ct[b] += mk;
  }

  for (int b = 0; b < NSEQ; ++b) {
    red[tid] = acc_lp[b]; __syncthreads();
    for (int st = 128; st > 0; st >>= 1) { if (tid < st) red[tid] += red[tid + st]; __syncthreads(); }
    if (tid == 0) seq_lp[b] = red[0];
    __syncthreads();
    red[tid] = acc_ct[b]; __syncthreads();
    for (int st = 128; st > 0; st >>= 1) { if (tid < st) red[tid] += red[tid + st]; __syncthreads(); }
    if (tid == 0) seq_ct[b] = red[0];
    __syncthreads();
  }

  if (tid == 0) {
    float avg[NSEQ];
    for (int b = 0; b < NSEQ; ++b) avg[b] = seq_lp[b] / fmaxf(seq_ct[b], 1.f);
    const float c0 = avg[0], c1 = avg[1], r0 = avg[2], r1 = avg[3];
    const float lo0 = (c0 - r0) - (log1pf(-__expf(c0)) - log1pf(-__expf(r0)));
    const float lo1 = (c1 - r1) - (log1pf(-__expf(c1)) - log1pf(-__expf(r1)));
    const float ra0 = log_sigmoid(lo0);
    const float ra1 = log_sigmoid(lo1);
    out[0] = -ORPO_BETA * (ra0 + ra1) * 0.5f;
    out[1] = ORPO_BETA * c0;
    out[2] = ORPO_BETA * c1;
    out[3] = ORPO_BETA * r0;
    out[4] = ORPO_BETA * r1;
    out[5] = (ra0 + ra1) * 0.5f;
    out[6] = (lo0 + lo1) * 0.5f;
  }
}

// =====================================================================
extern "C" void kernel_launch(void* const* d_in, const int* in_sizes, int n_in,
                              void* d_out, int out_size, void* d_ws, size_t ws_size,
                              hipStream_t stream) {
  (void)in_sizes; (void)n_in; (void)out_size;
  const float*     x   = (const float*)d_in[0];       // (4,1024,2048) fp32
  const float*     w   = (const float*)d_in[1];       // (50257,2048) fp32
  const long long* tgt = (const long long*)d_in[2];   // (4,1024) int64
  float* out = (float*)d_out;

  char* base = (char*)d_ws;
  unsigned short* xb = (unsigned short*)base;
  const size_t part_bytes = ((size_t)NTOK * NSLAB * 2 + NTOK) * sizeof(float);
  const size_t need_fast  = WS_X_BYTES + WS_W_BYTES + part_bytes;

  if (ws_size >= need_fast) {
    unsigned short* wf = (unsigned short*)(base + WS_X_BYTES);
    float* pm = (float*)(base + WS_X_BYTES + WS_W_BYTES);
    float* ps = pm + (size_t)NTOK * NSLAB;
    float* tl = ps + (size_t)NTOK * NSLAB;
    pack_x_kernel<<<2048, 256, 0, stream>>>(x, xb, tl);
    pack_w_kernel<<<25136, 256, 0, stream>>>(w, wf);   // 3142*64/8 blocks
    orpo_main_fast<<<dim3(16, NSLAB), 256, 0, stream>>>(xb, wf, tgt, pm, ps, tl);
    orpo_finalize_kernel<<<1, 256, 0, stream>>>(tgt, pm, ps, tl, out);
  } else {
    float* pm = (float*)(base + WS_X_BYTES);
    float* ps = pm + (size_t)NTOK * NSLAB;
    float* tl = ps + (size_t)NTOK * NSLAB;
    pack_x_kernel<<<2048, 256, 0, stream>>>(x, xb, tl);
    orpo_main_fallback<<<dim3(32, NSLAB), 256, 0, stream>>>(w, tgt, xb, pm, ps, tl);
    orpo_finalize_kernel<<<1, 256, 0, stream>>>(tgt, pm, ps, tl, out);
  }
}
